// MStAMPNNLayer_70334384439337
// MI455X (gfx1250) — matrix operand order repaired and checked by reading, not compiled, not run on hardware
//
#include <hip/hip_runtime.h>

// ---------------------------------------------------------------------------
// MStA-MPNN layer for MI455X (gfx1250, wave32, WMMA).
// All GEMMs (~95% of FLOPs) run on v_wmma_f32_16x16x32_bf16 with fp32
// accumulation; softmax / layernorm / segment-mean stay fp32.
// Each GEMM wave register-blocks a 64x32 output tile = 8 WMMAs per k-step.
// ---------------------------------------------------------------------------

typedef __attribute__((ext_vector_type(16))) __bf16        v16bf;
typedef __attribute__((ext_vector_type(8)))  float         v8f;
typedef __attribute__((ext_vector_type(8)))  unsigned int  v8u;

static constexpr int SS = 8;          // sequences
static constexpr int NN = 1024;       // nodes
static constexpr int EE = 16384;      // edges
static constexpr int HH = 128;        // hidden
static constexpr int CC = 32;         // head dim
static constexpr int NHEADS = 4;
static constexpr int MM = 7;          // S - 1

static constexpr long SN = (long)SS * NN;        // 8192 rows
static constexpr long SE = (long)SS * EE;        // 131072 rows

__device__ __forceinline__ unsigned short f2bf(float f) {
  unsigned int u = __builtin_bit_cast(unsigned int, f);
  unsigned int r = u + 0x7FFFu + ((u >> 16) & 1u);   // round-to-nearest-even
  return (unsigned short)(r >> 16);
}
__device__ __forceinline__ float bf2f(unsigned short b) {
  unsigned int u = ((unsigned int)b) << 16;
  return __builtin_bit_cast(float, u);
}

// ---------------------------------------------------------------------------
// Generic WMMA GEMM:  out = act( A[rows,K](bf16, optional row-gather)
//                                 @ W[cols, ldw](bf16, col-slice at koff)^T
//                                 + accin + bias )
// Wave computes a 64x32 tile (4 M x 2 N subtiles); K-loop in steps of 32.
// Fragment layouts per CDNA5 ISA 7.12.2 (wave32).
// Requires rows % 64 == 0, cols % 32 == 0, K % 32 == 0 (true for all uses).
// ---------------------------------------------------------------------------
__global__ __launch_bounds__(256) void gemm_wmma_bf16(
    const unsigned short* __restrict__ A, int lda, const int* __restrict__ rowmap,
    const unsigned short* __restrict__ W, int ldw, int koff, int K,
    const float* __restrict__ bias, const float* __restrict__ accin,
    float* __restrict__ outf, unsigned short* __restrict__ outb,
    int rows, int cols, int relu)
{
  const int lane = threadIdx.x & 31;
  const int wid  = threadIdx.x >> 5;
  const int Nt = cols >> 5;                  // 32-wide N blocks
  const int Mt = rows >> 6;                  // 64-tall M blocks
  const int gid = blockIdx.x * 8 + wid;
  if (gid >= Mt * Nt) return;
  const int tn = gid % Nt, tm = gid / Nt;
  const int grp = lane >> 4, l15 = lane & 15;

  const unsigned int* ap[4];
#pragma unroll
  for (int i = 0; i < 4; ++i) {
    int r = tm * 64 + i * 16 + l15;
    if (rowmap) r = rowmap[r];
    ap[i] = (const unsigned int*)A + (((long)r * lda) >> 1);
  }
  const unsigned int* wp[2];
#pragma unroll
  for (int j = 0; j < 2; ++j)
    wp[j] = (const unsigned int*)W +
            ((((long)(tn * 32 + j * 16 + l15)) * ldw + koff) >> 1);

  v8f c[4][2];
  if (accin) {
#pragma unroll
    for (int i = 0; i < 4; ++i)
#pragma unroll
      for (int j = 0; j < 2; ++j) {
        const float* ai = accin +
            ((long)(tm * 64 + i * 16 + 8 * grp)) * cols + tn * 32 + j * 16 + l15;
#pragma unroll
        for (int r = 0; r < 8; ++r) c[i][j][r] = ai[(long)r * cols];
      }
  } else {
#pragma unroll
    for (int i = 0; i < 4; ++i)
#pragma unroll
      for (int j = 0; j < 2; ++j)
#pragma unroll
        for (int r = 0; r < 8; ++r) c[i][j][r] = 0.f;
  }

  for (int k0 = 0; k0 < K; k0 += 32) {
    const int ab = k0 >> 1;                  // u32 index base within row
    v16bf a[4], b[2];
#pragma unroll
    for (int i = 0; i < 4; ++i) {
      v8u au;
#pragma unroll
      for (int q = 0; q < 8; ++q) {
        // A (16x32, MxK): halves 2q,2q+1 -> k = 2q + (q>=4?8:0) + 8*grp
        const int pk = (q < 4) ? (q + 4 * grp) : (q + 4 + 4 * grp);
        au[q] = ap[i][ab + pk];
      }
      a[i] = __builtin_bit_cast(v16bf, au);
    }
#pragma unroll
    for (int j = 0; j < 2; ++j) {
      v8u bu;
#pragma unroll
      for (int q = 0; q < 8; ++q) bu[q] = wp[j][ab + ((q < 4) ? (q + 4 * grp) : (q + 4 + 4 * grp))];
      b[j] = __builtin_bit_cast(v16bf, bu);
    }
    // prefetch next k-step's A fragments (speculative; tail overrun dropped)
#pragma unroll
    for (int i = 0; i < 4; ++i)
      __builtin_prefetch((const void*)(ap[i] + ab + 16), 0, 3);
#pragma unroll
    for (int i = 0; i < 4; ++i)
#pragma unroll
      for (int j = 0; j < 2; ++j)
        c[i][j] = __builtin_amdgcn_wmma_f32_16x16x32_bf16(
            false, a[i], false, b[j], (short)0, c[i][j], false, false);
  }

#pragma unroll
  for (int j = 0; j < 2; ++j) {
    const int col = tn * 32 + j * 16 + l15;
    const float bv = bias ? bias[col] : 0.f;
#pragma unroll
    for (int i = 0; i < 4; ++i) {
#pragma unroll
      for (int r = 0; r < 8; ++r) {
        const int row = tm * 64 + i * 16 + r + 8 * grp;
        float v = c[i][j][r] + bv;
        if (relu) v = fmaxf(v, 0.f);
        const long o = (long)row * cols + col;
        if (outf) outf[o] = v;
        if (outb) outb[o] = f2bf(v);
      }
    }
  }
}

// --------------------------- elementwise helpers ---------------------------
__global__ void cvt_f32_bf16(const float* __restrict__ in,
                             unsigned short* __restrict__ out, long n) {
  long i = (long)blockIdx.x * blockDim.x + threadIdx.x;
  if (i < n) out[i] = f2bf(in[i]);
}

__global__ void zero_f32(float* __restrict__ p, long n) {
  long i = (long)blockIdx.x * blockDim.x + threadIdx.x;
  if (i < n) p[i] = 0.f;
}

// rowmap[s*E+e] = s*N + edge_idx[s, which, e]
__global__ void build_rowmaps(const int* __restrict__ eidx,
                              int* __restrict__ rsrc, int* __restrict__ rtgt) {
  long i = (long)blockIdx.x * blockDim.x + threadIdx.x;
  if (i >= SE) return;
  int e = (int)(i % EE), s = (int)(i / EE);
  rsrc[i] = s * NN + eidx[((long)s * 2 + 0) * EE + e];
  rtgt[i] = s * NN + eidx[((long)s * 2 + 1) * EE + e];
}

// ------------------------- _place mapping (scan) ---------------------------
__global__ __launch_bounds__(1024) void scan_mask_kernel(
    const unsigned char* __restrict__ mask, int L, int* __restrict__ rank) {
  __shared__ int sm[1024];
  __shared__ int carry;
  const int tid = threadIdx.x;
  if (tid == 0) carry = 0;
  __syncthreads();
  for (int base = 0; base < L; base += 1024) {
    int i = base + tid;
    int v = (i < L) ? (mask[i] ? 1 : 0) : 0;
    sm[tid] = v;
    __syncthreads();
    for (int s = 1; s < 1024; s <<= 1) {           // Hillis-Steele inclusive
      int t = (tid >= s) ? sm[tid - s] : 0;
      __syncthreads();
      sm[tid] += t;
      __syncthreads();
    }
    if (i < L) rank[i] = carry + sm[tid] - v;      // exclusive rank
    __syncthreads();
    if (tid == 1023) carry += sm[1023];
    __syncthreads();
  }
}

__global__ void build_pos_kernel(const unsigned char* __restrict__ mask,
                                 const int* __restrict__ rank,
                                 int* __restrict__ pos, int L) {
  int i = blockIdx.x * blockDim.x + threadIdx.x;
  if (i < L && mask[i]) pos[rank[i]] = i;
}

__global__ void build_map_kernel(const unsigned char* __restrict__ tmask,
                                 const int* __restrict__ trank,
                                 const int* __restrict__ pos,
                                 int* __restrict__ map, int L) {
  int i = blockIdx.x * blockDim.x + threadIdx.x;
  if (i < L) map[i] = tmask[i] ? pos[trank[i]] : -1;
}

// amask[0,n] = any_m tmask[m,n];  amask[1+m,n] = tmask[m,n]
__global__ void build_amask_kernel(const unsigned char* __restrict__ tmask,
                                   float* __restrict__ amask, int Nn) {
  int n = blockIdx.x * blockDim.x + threadIdx.x;
  if (n >= Nn) return;
  int any = 0;
  for (int m = 0; m < MM; ++m) {
    int t = tmask[(long)m * Nn + n] ? 1 : 0;
    amask[(long)(1 + m) * Nn + n] = (float)t;
    any |= t;
  }
  amask[n] = (float)any;
}

// attn_h: row0 = h[0]; row(1+m,n) = map>=0 ? h[1 + src_m, src_n] : 0   (bf16)
__global__ void build_attnh_kernel(const float* __restrict__ hw,
                                   const int* __restrict__ map,
                                   unsigned short* __restrict__ AH, int Nn) {
  long idx = (long)blockIdx.x * blockDim.x + threadIdx.x;
  const long total = (long)SS * Nn * HH;
  if (idx >= total) return;
  int h = (int)(idx % HH);
  long r = idx / HH;
  int n = (int)(r % Nn);
  int s = (int)(r / Nn);
  float v;
  if (s == 0) {
    v = hw[(long)n * HH + h];
  } else {
    int p = map[(long)(s - 1) * Nn + n];
    v = (p < 0) ? 0.f
                : hw[((long)(1 + p / Nn) * Nn + (p % Nn)) * HH + h];
  }
  AH[idx] = f2bf(v);
}

// ------------------------- message segment-mean ----------------------------
__global__ void segsum_num_kernel(const float* __restrict__ msg,
                                  const int* __restrict__ eidx,
                                  const float* __restrict__ emask,
                                  float* __restrict__ num) {
  long idx = (long)blockIdx.x * blockDim.x + threadIdx.x;
  const long total = SE * HH;
  if (idx >= total) return;
  int h = (int)(idx % HH);
  long se = idx / HH;
  int e = (int)(se % EE), s = (int)(se / EE);
  float m = emask[(long)s * EE + e];
  if (m == 0.f) return;
  int t = eidx[((long)s * 2 + 1) * EE + e];
  atomicAdd(&num[((long)s * NN + t) * HH + h], msg[idx] * m);
}

__global__ void segsum_den_kernel(const int* __restrict__ eidx,
                                  const float* __restrict__ emask,
                                  float* __restrict__ den) {
  long i = (long)blockIdx.x * blockDim.x + threadIdx.x;
  if (i >= SE) return;
  int e = (int)(i % EE), s = (int)(i / EE);
  float m = emask[i];
  if (m == 0.f) return;
  int t = eidx[((long)s * 2 + 1) * EE + e];
  atomicAdd(&den[(long)s * NN + t], m);
}

__global__ void div_den_kernel(float* __restrict__ num,
                               const float* __restrict__ den) {
  long idx = (long)blockIdx.x * blockDim.x + threadIdx.x;
  const long total = SN * HH;
  if (idx >= total) return;
  num[idx] = num[idx] / fmaxf(den[idx / HH], 1.f);
}

// ------------------------------- layernorm ---------------------------------
__global__ __launch_bounds__(128) void ln_kernel(
    const float* __restrict__ base, const float* __restrict__ delta,
    const float* __restrict__ g, const float* __restrict__ bb,
    float* __restrict__ out) {
  __shared__ float red[128];
  const int h = threadIdx.x;
  const long o = (long)blockIdx.x * HH + h;
  float x = base[o] + (delta ? delta[o] : 0.f);
  red[h] = x;
  __syncthreads();
  for (int s = 64; s > 0; s >>= 1) { if (h < s) red[h] += red[h + s]; __syncthreads(); }
  const float mean = red[0] / (float)HH;
  __syncthreads();
  const float d = x - mean;
  red[h] = d * d;
  __syncthreads();
  for (int s = 64; s > 0; s >>= 1) { if (h < s) red[h] += red[h + s]; __syncthreads(); }
  const float inv = rsqrtf(red[0] / (float)HH + 1e-5f);
  out[o] = d * inv * g[h] + bb[h];
}

// -------------------- tiny S x S attention per (n, head) -------------------
__global__ void attn_combine_kernel(const unsigned short* __restrict__ Qb,
                                    const unsigned short* __restrict__ Kb,
                                    const unsigned short* __restrict__ Vb,
                                    const float* __restrict__ amask,
                                    unsigned short* __restrict__ OB, int Nn) {
  long idx = (long)blockIdx.x * blockDim.x + threadIdx.x;
  const long total = (long)SS * Nn * NHEADS;
  if (idx >= total) return;
  const int head = (int)(idx % NHEADS);
  const int n = (int)((idx / NHEADS) % Nn);
  const int a = (int)(idx / ((long)NHEADS * Nn));
  if (amask[(long)a * Nn + n] == 0.f) return;     // keep h@Wv prefill
  const float scale = 0.17677669529663687f;       // 1/sqrt(32)
  const long qo = ((long)a * Nn + n) * HH + head * CC;
  float q[CC];
#pragma unroll
  for (int c = 0; c < CC; ++c) q[c] = bf2f(Qb[qo + c]);
  float sc[SS];
  float mx = -1e30f;
#pragma unroll
  for (int b = 0; b < SS; ++b) {
    const long ko = ((long)b * Nn + n) * HH + head * CC;
    float s = 0.f;
#pragma unroll
    for (int c = 0; c < CC; ++c) s += q[c] * bf2f(Kb[ko + c]);
    s = s * scale - (1.f - amask[(long)b * Nn + n]) * 1e9f;
    sc[b] = s;
    mx = fmaxf(mx, s);
  }
  float den = 0.f;
#pragma unroll
  for (int b = 0; b < SS; ++b) { sc[b] = expf(sc[b] - mx); den += sc[b]; }
  const float inv = 1.f / den;
  float o[CC];
#pragma unroll
  for (int c = 0; c < CC; ++c) o[c] = 0.f;
#pragma unroll
  for (int b = 0; b < SS; ++b) {
    const float p = sc[b] * inv;
    const long vo = ((long)b * Nn + n) * HH + head * CC;
#pragma unroll
    for (int c = 0; c < CC; ++c) o[c] += p * bf2f(Vb[vo + c]);
  }
#pragma unroll
  for (int c = 0; c < CC; ++c) OB[qo + c] = f2bf(o[c]);
}

// ---------------------------------------------------------------------------
extern "C" void kernel_launch(void* const* d_in, const int* in_sizes, int n_in,
                              void* d_out, int out_size, void* d_ws, size_t ws_size,
                              hipStream_t stream) {
  (void)in_sizes; (void)n_in; (void)out_size; (void)ws_size;

  // ---- inputs (setup_inputs() dict order; params dict flattened in order) --
  const float* h_V  = (const float*)d_in[0];
  const float* h_E  = (const float*)d_in[1];
  const int*   eidx = (const int*)d_in[2];
  const float* emask = (const float*)d_in[3];
  const unsigned char* msta_mask   = (const unsigned char*)d_in[4];
  const unsigned char* msta_emask  = (const unsigned char*)d_in[5];
  const unsigned char* tgt_mask    = (const unsigned char*)d_in[6];
  const unsigned char* tgt_emask   = (const unsigned char*)d_in[7];
  auto F = [&](int i) { return (const float*)d_in[i]; };

  float* outV = (float*)d_out;                 // S*N*H
  float* outE = outV + SN * HH;                // S*E*H

  // ---- workspace carve ----------------------------------------------------
  char* base = (char*)d_ws;
  size_t off = 0;
  auto carve = [&](size_t bytes) -> char* {
    char* p = base + off;
    off = (off + bytes + 255) & ~(size_t)255;
    return p;
  };
  auto carveU16 = [&](long n) { return (unsigned short*)carve((size_t)n * 2); };
  auto carveF32 = [&](long n) { return (float*)carve((size_t)n * 4); };
  auto carveI32 = [&](long n) { return (int*)carve((size_t)n * 4); };

  // bf16 weights
  unsigned short* wb_nm0 = carveU16(128L * 384);
  unsigned short* wb_nm1 = carveU16(128L * 128);
  unsigned short* wb_nm2 = carveU16(128L * 128);
  unsigned short* wb_nd0 = carveU16(512L * 128);
  unsigned short* wb_nd1 = carveU16(128L * 512);
  unsigned short* wb_em0 = carveU16(128L * 384);
  unsigned short* wb_em1 = carveU16(128L * 128);
  unsigned short* wb_em2 = carveU16(128L * 128);
  unsigned short* wb_ed0 = carveU16(512L * 128);
  unsigned short* wb_ed1 = carveU16(128L * 512);
  unsigned short* wb_nq = carveU16(128L * 128);
  unsigned short* wb_nk = carveU16(128L * 128);
  unsigned short* wb_nv = carveU16(128L * 128);
  unsigned short* wb_no = carveU16(128L * 128);
  unsigned short* wb_eq = carveU16(128L * 128);
  unsigned short* wb_ek = carveU16(128L * 128);
  unsigned short* wb_ev = carveU16(128L * 128);
  unsigned short* wb_eo = carveU16(128L * 128);

  int* rmap_src = carveI32(SE);
  int* rmap_tgt = carveI32(SE);
  int* rank1    = carveI32((long)MM * EE);
  int* rank2    = carveI32((long)MM * EE);
  int* posbuf   = carveI32((long)MM * EE);
  int* map_node = carveI32((long)MM * NN);
  int* map_edge = carveI32((long)MM * EE);
  float* am_node = carveF32(SN);
  float* am_edge = carveF32(SE);

  unsigned short* hVbf = carveU16(SN * HH);
  unsigned short* hEbf = carveU16(SE * HH);
  unsigned short* AH   = carveU16(SE * HH);
  // 128 MB region time-shared: dense hidden (SE x 512 bf16) OR Q/K/V/O (bf16)
  unsigned short* R1 = carveU16(SE * 512);
  unsigned short* Qb = R1;
  unsigned short* Kb = R1 + SE * HH;
  unsigned short* Vb = R1 + 2 * SE * HH;
  unsigned short* OB = R1 + 3 * SE * HH;
  unsigned short* X1a = R1;              // mlp hidden 1 (SE x 128 bf16)
  unsigned short* X1b = R1 + SE * HH;    // mlp hidden 2

  float* MSG = carveF32(SE * HH);        // fp32 GEMM epilogue buffer (64 MB)
  float* hVw = carveF32(SN * HH);
  float* hEw = carveF32(SE * HH);
  float* NUM = carveF32(SN * HH);
  float* DEN = carveF32(SN);

  // ---- launch helpers -----------------------------------------------------
  auto cvt = [&](const float* in, unsigned short* out, long n) {
    cvt_f32_bf16<<<(int)((n + 255) / 256), 256, 0, stream>>>(in, out, n);
  };
  auto zero = [&](float* p, long n) {
    zero_f32<<<(int)((n + 255) / 256), 256, 0, stream>>>(p, n);
  };
  auto gemm = [&](const unsigned short* A, int lda, const int* rowmap,
                  const unsigned short* W, int ldw, int koff, int K,
                  const float* bias, const float* accin,
                  float* outf, unsigned short* outb, long rows, int cols, int relu) {
    long tiles = (rows / 64) * (cols / 32);
    gemm_wmma_bf16<<<(int)((tiles + 7) / 8), 256, 0, stream>>>(
        A, lda, rowmap, W, ldw, koff, K, bias, accin, outf, outb,
        (int)rows, cols, relu);
  };
  auto ln = [&](const float* b0, const float* dl, const float* g,
                const float* bt, float* out, long rows) {
    ln_kernel<<<(int)rows, 128, 0, stream>>>(b0, dl, g, bt, out);
  };

  // ---- phase 0: weight conversion, rowmaps, place-maps --------------------
  struct WCvt { int idx; unsigned short* dst; long n; };
  const WCvt wl[] = {
      {8, wb_nm0, 128L * 384}, {10, wb_nm1, 128L * 128}, {12, wb_nm2, 128L * 128},
      {14, wb_nd0, 512L * 128}, {16, wb_nd1, 128L * 512},
      {18, wb_em0, 128L * 384}, {20, wb_em1, 128L * 128}, {22, wb_em2, 128L * 128},
      {24, wb_ed0, 512L * 128}, {26, wb_ed1, 128L * 512},
      {28, wb_nq, 128L * 128}, {29, wb_nk, 128L * 128},
      {30, wb_nv, 128L * 128}, {31, wb_no, 128L * 128},
      {35, wb_eq, 128L * 128}, {36, wb_ek, 128L * 128},
      {37, wb_ev, 128L * 128}, {38, wb_eo, 128L * 128}};
  for (const auto& wc : wl) cvt(F(wc.idx), wc.dst, wc.n);

  build_rowmaps<<<(int)((SE + 255) / 256), 256, 0, stream>>>(eidx, rmap_src, rmap_tgt);

  // node place mapping
  {
    const int L = MM * NN;
    scan_mask_kernel<<<1, 1024, 0, stream>>>(msta_mask, L, rank1);
    scan_mask_kernel<<<1, 1024, 0, stream>>>(tgt_mask, L, rank2);
    build_pos_kernel<<<(L + 255) / 256, 256, 0, stream>>>(msta_mask, rank1, posbuf, L);
    build_map_kernel<<<(L + 255) / 256, 256, 0, stream>>>(tgt_mask, rank2, posbuf, map_node, L);
    build_amask_kernel<<<(NN + 255) / 256, 256, 0, stream>>>(tgt_mask, am_node, NN);
  }
  // edge place mapping
  {
    const int L = MM * EE;
    scan_mask_kernel<<<1, 1024, 0, stream>>>(msta_emask, L, rank1);
    scan_mask_kernel<<<1, 1024, 0, stream>>>(tgt_emask, L, rank2);
    build_pos_kernel<<<(L + 255) / 256, 256, 0, stream>>>(msta_emask, rank1, posbuf, L);
    build_map_kernel<<<(L + 255) / 256, 256, 0, stream>>>(tgt_emask, rank2, posbuf, map_edge, L);
    build_amask_kernel<<<(EE + 255) / 256, 256, 0, stream>>>(tgt_emask, am_edge, EE);
  }

  cvt(h_V, hVbf, SN * HH);
  cvt(h_E, hEbf, SE * HH);

  // ---- phase 1: node message passing --------------------------------------
  // msg = mlp3(concat[h_E, hV[src], hV[tgt]]) as 3 accumulating K=128 GEMMs
  gemm(hEbf, HH, nullptr, wb_nm0, 384, 0, 128, nullptr, nullptr, MSG, nullptr, SE, HH, 0);
  gemm(hVbf, HH, rmap_src, wb_nm0, 384, 128, 128, nullptr, MSG, MSG, nullptr, SE, HH, 0);
  gemm(hVbf, HH, rmap_tgt, wb_nm0, 384, 256, 128, F(9), MSG, nullptr, X1a, SE, HH, 1);
  gemm(X1a, HH, nullptr, wb_nm1, 128, 0, 128, F(11), nullptr, nullptr, X1b, SE, HH, 1);
  gemm(X1b, HH, nullptr, wb_nm2, 128, 0, 128, F(13), nullptr, MSG, nullptr, SE, HH, 0);

  zero(NUM, SN * HH);
  zero(DEN, SN);
  segsum_num_kernel<<<(int)((SE * HH + 255) / 256), 256, 0, stream>>>(MSG, eidx, emask, NUM);
  segsum_den_kernel<<<(int)((SE + 255) / 256), 256, 0, stream>>>(eidx, emask, DEN);
  div_den_kernel<<<(int)((SN * HH + 255) / 256), 256, 0, stream>>>(NUM, DEN);
  ln(h_V, NUM, F(42), F(43), hVw, SN);                       // node_norm0

  // ---- phase 2: node MStA attention ---------------------------------------
  cvt(hVw, hVbf, SN * HH);
  build_attnh_kernel<<<(int)((SN * HH + 255) / 256), 256, 0, stream>>>(hVw, map_node, AH, NN);
  gemm(AH, HH, nullptr, wb_nq, 128, 0, 128, nullptr, nullptr, nullptr, Qb, SN, HH, 0);
  gemm(AH, HH, nullptr, wb_nk, 128, 0, 128, nullptr, nullptr, nullptr, Kb, SN, HH, 0);
  gemm(AH, HH, nullptr, wb_nv, 128, 0, 128, nullptr, nullptr, nullptr, Vb, SN, HH, 0);
  gemm(hVbf, HH, nullptr, wb_nv, 128, 0, 128, nullptr, nullptr, nullptr, OB, SN, HH, 0); // h@Wv prefill
  attn_combine_kernel<<<(int)((SN * NHEADS + 255) / 256), 256, 0, stream>>>(
      Qb, Kb, Vb, am_node, OB, NN);
  gemm(OB, HH, nullptr, wb_no, 128, 0, 128, F(32), nullptr, MSG, nullptr, SN, HH, 0);
  ln(hVw, MSG, F(33), F(34), hVw, SN);                       // node_attn LN

  // ---- phase 3: node dense FFN --------------------------------------------
  cvt(hVw, hVbf, SN * HH);
  gemm(hVbf, HH, nullptr, wb_nd0, 128, 0, 128, F(15), nullptr, nullptr, R1, SN, 512, 1);
  gemm(R1, 512, nullptr, wb_nd1, 512, 0, 512, F(17), nullptr, MSG, nullptr, SN, HH, 0);
  ln(hVw, MSG, F(44), F(45), outV, SN);                      // node_norm1 -> output h_V

  // ---- phase 4: edge MLP ---------------------------------------------------
  cvt(outV, hVbf, SN * HH);                                  // updated h_V
  gemm(hEbf, HH, nullptr, wb_em0, 384, 0, 128, nullptr, nullptr, MSG, nullptr, SE, HH, 0);
  gemm(hVbf, HH, rmap_src, wb_em0, 384, 128, 128, nullptr, MSG, MSG, nullptr, SE, HH, 0);
  gemm(hVbf, HH, rmap_tgt, wb_em0, 384, 256, 128, F(19), MSG, nullptr, X1a, SE, HH, 1);
  gemm(X1a, HH, nullptr, wb_em1, 128, 0, 128, F(21), nullptr, nullptr, X1b, SE, HH, 1);
  gemm(X1b, HH, nullptr, wb_em2, 128, 0, 128, F(23), nullptr, MSG, nullptr, SE, HH, 0);
  ln(h_E, MSG, F(46), F(47), hEw, SE);                       // edge_norm0

  // ---- phase 5: edge MStA attention ---------------------------------------
  cvt(hEw, hEbf, SE * HH);
  build_attnh_kernel<<<(int)((SE * HH + 255) / 256), 256, 0, stream>>>(hEw, map_edge, AH, EE);
  gemm(AH, HH, nullptr, wb_eq, 128, 0, 128, nullptr, nullptr, nullptr, Qb, SE, HH, 0);
  gemm(AH, HH, nullptr, wb_ek, 128, 0, 128, nullptr, nullptr, nullptr, Kb, SE, HH, 0);
  gemm(AH, HH, nullptr, wb_ev, 128, 0, 128, nullptr, nullptr, nullptr, Vb, SE, HH, 0);
  gemm(hEbf, HH, nullptr, wb_ev, 128, 0, 128, nullptr, nullptr, nullptr, OB, SE, HH, 0);
  attn_combine_kernel<<<(int)((SE * NHEADS + 255) / 256), 256, 0, stream>>>(
      Qb, Kb, Vb, am_edge, OB, EE);
  gemm(OB, HH, nullptr, wb_eo, 128, 0, 128, F(39), nullptr, MSG, nullptr, SE, HH, 0);
  ln(hEw, MSG, F(40), F(41), hEw, SE);                       // edge_attn LN

  // ---- phase 6: edge dense FFN --------------------------------------------
  cvt(hEw, hEbf, SE * HH);
  gemm(hEbf, HH, nullptr, wb_ed0, 128, 0, 128, F(25), nullptr, nullptr, R1, SE, 512, 1);
  gemm(R1, 512, nullptr, wb_ed1, 512, 0, 512, F(27), nullptr, MSG, nullptr, SE, HH, 0);
  ln(hEw, MSG, F(48), F(49), outE, SE);                      // edge_norm1 -> output h_E
}
